// TalosJEPA_38036230373782
// MI455X (gfx1250) — compile-verified, matrix operand order checked by reading, not run on hardware
//
#include <hip/hip_runtime.h>
#include <hip/hip_bf16.h>
#include <stdint.h>

// ---------------------------------------------------------------------------
// MI455X (gfx1250) implementation of the Talos-JEPA forward pass.
// All GEMMs run on v_wmma_f32_16x16x32_bf16 (f32 accumulate), activations and
// weights are staged as bf16. Global->LDS uses the CDNA5 async-to-LDS path
// (ASYNCcnt) when the toolchain exposes the builtin.
// ---------------------------------------------------------------------------

#define D_DIM 1024
#define BM 128
#define BN 128
#define BK 64
#define KB_STEPS (D_DIM / BK) /* 16 */

typedef __attribute__((ext_vector_type(16))) __bf16 v16bf;
typedef __attribute__((ext_vector_type(4)))  __bf16 v4bf;
typedef __attribute__((ext_vector_type(8)))  float  v8f;

struct Frag {
  union {
    v16bf v;
    unsigned u[8];
  };
};

#if defined(__gfx1250__) && __has_builtin(__builtin_amdgcn_global_load_async_to_lds_b128)
#define USE_ASYNC_LDS 1
#else
#define USE_ASYNC_LDS 0
#endif

#if USE_ASYNC_LDS
// Exact parameter types per hipcc diagnostic: pointee is a GCC-vector
// 'int __attribute__((vector_size(16)))'; global side lives in the device
// (addrspace(1)) AS, LDS side in addrspace(3).
typedef int v4i_gcc __attribute__((vector_size(16)));
typedef __attribute__((address_space(1))) v4i_gcc* g_v4i_ptr;
typedef __attribute__((address_space(3))) v4i_gcc* l_v4i_ptr;
#endif

__device__ __forceinline__ void copy16_to_lds(void* lds, const void* g) {
#if USE_ASYNC_LDS
  __builtin_amdgcn_global_load_async_to_lds_b128(
      (g_v4i_ptr)(uintptr_t)g, (l_v4i_ptr)(uintptr_t)lds,
      /*offset=*/0, /*cpol=*/0);
#else
  *(int4*)lds = *(const int4*)g;
#endif
}

__device__ __forceinline__ void wait_async_pending8() {
#if USE_ASYNC_LDS
#if __has_builtin(__builtin_amdgcn_s_wait_asynccnt)
  __builtin_amdgcn_s_wait_asynccnt(8);
#else
  asm volatile("s_wait_asynccnt 8" ::: "memory");
#endif
#endif
}

__device__ __forceinline__ void wait_async_all() {
#if USE_ASYNC_LDS
#if __has_builtin(__builtin_amdgcn_s_wait_asynccnt)
  __builtin_amdgcn_s_wait_asynccnt(0);
#else
  asm volatile("s_wait_asynccnt 0" ::: "memory");
#endif
#endif
}

// Load one 16x32 bf16 fragment per the CDNA5 16-bit A layout:
// lane L (L<16): row M=L, K = {0..7}u{16..23}; lane L+16: K = {8..15}u{24..31}.
// Each VGPR holds a contiguous K pair, so 8 x b32 LDS loads per fragment.
// p must already point at &row[kk + 8*half].
__device__ __forceinline__ void ldfrag(Frag& f, const __bf16* p) {
  const unsigned* q = (const unsigned*)p;
  f.u[0] = q[0];
  f.u[1] = q[1];
  f.u[2] = q[2];
  f.u[3] = q[3];
  f.u[4] = q[8];
  f.u[5] = q[9];
  f.u[6] = q[10];
  f.u[7] = q[11];
}

// ---------------------------------------------------------------------------
// Fused GEMM: C = A[16384,1024] * W^T (+bias), epilogue per MODE:
//   MODE 0 (GATE):   out_bf16 = sigmoid(C) * x * decay            (new_state)
//   MODE 1 (OUT):    out_f32  = C + x                             (pre-LN)
//   MODE 2 (EXPERT): pred    (+)= gate_prob[n,e] * C
// ---------------------------------------------------------------------------
template <int MODE>
__global__ __launch_bounds__(256) void liquid_gemm(
    const __bf16* __restrict__ A, const __bf16* __restrict__ W,
    const float* __restrict__ bias, const float* __restrict__ aux,
    const __bf16* __restrict__ X, const float* __restrict__ gp, int expert_e,
    int accum, __bf16* __restrict__ outB, float* __restrict__ outF) {
  __shared__ __align__(16) __bf16 lA[2][BM * BK];
  __shared__ __align__(16) __bf16 lB[2][BN * BK];

  const int tid = threadIdx.x;
  const int lane = tid & 31;
  const int wid = tid >> 5;
  const int wm = wid & 1;   // 2 wave rows   -> 64 M each
  const int wn = wid >> 1;  // 4 wave cols   -> 32 N each
  const int m0 = blockIdx.y * BM;
  const int n0 = blockIdx.x * BN;
  const int half = lane >> 4;
  const int l16 = lane & 15;

  v8f acc[4][2];
#pragma unroll
  for (int mt = 0; mt < 4; ++mt)
#pragma unroll
    for (int nt = 0; nt < 2; ++nt)
#pragma unroll
      for (int r = 0; r < 8; ++r) acc[mt][nt][r] = 0.f;

  auto stage = [&](int kb, int buf) {
    const int k0 = kb * BK;
#pragma unroll
    for (int i = 0; i < 4; ++i) {
      const int c = i * 256 + tid;       // 1024 chunks of 16B per tile
      const int r = c >> 3;              // row within tile
      const int c8 = (c & 7) * 8;        // bf16 column within row
      copy16_to_lds(&lA[buf][r * BK + c8],
                    A + (size_t)(m0 + r) * D_DIM + k0 + c8);
      copy16_to_lds(&lB[buf][r * BK + c8],
                    W + (size_t)(n0 + r) * D_DIM + k0 + c8);
    }
  };

  stage(0, 0);
  for (int kb = 0; kb < KB_STEPS; ++kb) {
    const int buf = kb & 1;
    if (kb + 1 < KB_STEPS) {
      stage(kb + 1, buf ^ 1);  // 8 async ops in flight for next stage
      wait_async_pending8();   // current stage complete
    } else {
      wait_async_all();
    }
    __syncthreads();

    const __bf16* As = lA[buf];
    const __bf16* Bs = lB[buf];
#pragma unroll
    for (int kk = 0; kk < BK; kk += 32) {
      Frag a[4], b[2];
#pragma unroll
      for (int mt = 0; mt < 4; ++mt)
        ldfrag(a[mt], As + (wm * 64 + mt * 16 + l16) * BK + kk + half * 8);
#pragma unroll
      for (int nt = 0; nt < 2; ++nt)
        ldfrag(b[nt], Bs + (wn * 32 + nt * 16 + l16) * BK + kk + half * 8);
#pragma unroll
      for (int mt = 0; mt < 4; ++mt)
#pragma unroll
        for (int nt = 0; nt < 2; ++nt)
          acc[mt][nt] = __builtin_amdgcn_wmma_f32_16x16x32_bf16(
              false, a[mt].v, false, b[nt].v, (short)0, acc[mt][nt], false,
              false);
    }
    __syncthreads();  // all waves done reading before buffer is reused
  }

  // Epilogue. C/D layout: lane l holds column n = l&15; VGPR r is row
  // m = r + 8*(l>>4) within the 16x16 tile.
#pragma unroll
  for (int mt = 0; mt < 4; ++mt) {
#pragma unroll
    for (int nt = 0; nt < 2; ++nt) {
      const int gn = n0 + wn * 32 + nt * 16 + l16;
#pragma unroll
      for (int r = 0; r < 8; ++r) {
        const int gm = m0 + wm * 64 + mt * 16 + half * 8 + r;
        const size_t idx = (size_t)gm * D_DIM + gn;
        const float c = acc[mt][nt][r] + bias[gn];
        if (MODE == 0) {
          const float g = 1.f / (1.f + __expf(-c));
          outB[idx] = (__bf16)(g * (float)X[idx] * aux[gn]);
        } else if (MODE == 1) {
          outF[idx] = c + (float)X[idx];
        } else {
          const float o = gp[(size_t)gm * 4 + expert_e] * c;
          if (accum)
            outF[idx] += o;
          else
            outF[idx] = o;
        }
      }
    }
  }
}

// ---------------------------------------------------------------------------
// Row LayerNorm over D=1024; one 256-thread block (8 waves) per row.
// Writes bf16 (next-layer activations) and optionally f32 (final outputs).
// ---------------------------------------------------------------------------
__global__ __launch_bounds__(256) void ln_kernel(
    const float* __restrict__ in, const float* __restrict__ gamma,
    const float* __restrict__ beta, __bf16* __restrict__ outB,
    float* __restrict__ outF) {
  const int row = blockIdx.x;
  const int tid = threadIdx.x;
  const float4 v = ((const float4*)(in + (size_t)row * D_DIM))[tid];
  float s = v.x + v.y + v.z + v.w;
  float s2 = v.x * v.x + v.y * v.y + v.z * v.z + v.w * v.w;
#pragma unroll
  for (int o = 16; o >= 1; o >>= 1) {
    s += __shfl_xor(s, o, 32);
    s2 += __shfl_xor(s2, o, 32);
  }
  __shared__ float rs[8], rs2[8];
  if ((tid & 31) == 0) {
    rs[tid >> 5] = s;
    rs2[tid >> 5] = s2;
  }
  __syncthreads();
  s = 0.f;
  s2 = 0.f;
#pragma unroll
  for (int i = 0; i < 8; ++i) {
    s += rs[i];
    s2 += rs2[i];
  }
  const float mu = s * (1.f / (float)D_DIM);
  const float var = s2 * (1.f / (float)D_DIM) - mu * mu;
  const float rstd = rsqrtf(var + 1e-5f);
  const int c0 = tid * 4;
  const size_t base = (size_t)row * D_DIM + c0;
  float o0 = (v.x - mu) * rstd * gamma[c0 + 0] + beta[c0 + 0];
  float o1 = (v.y - mu) * rstd * gamma[c0 + 1] + beta[c0 + 1];
  float o2 = (v.z - mu) * rstd * gamma[c0 + 2] + beta[c0 + 2];
  float o3 = (v.w - mu) * rstd * gamma[c0 + 3] + beta[c0 + 3];
  v4bf ob = {(__bf16)o0, (__bf16)o1, (__bf16)o2, (__bf16)o3};
  *(v4bf*)(outB + base) = ob;
  if (outF) {
    float4 of = {o0, o1, o2, o3};
    *(float4*)(outF + base) = of;
  }
}

// ---------------------------------------------------------------------------
// Gate softmax over E=4 experts; one wave per token.
// ---------------------------------------------------------------------------
__global__ __launch_bounds__(256) void gate_kernel(
    const __bf16* __restrict__ z, const float* __restrict__ gW,
    const float* __restrict__ gb, float* __restrict__ gp, int ntok) {
  const int lane = threadIdx.x & 31;
  const int wid = threadIdx.x >> 5;
  const int t = blockIdx.x * 8 + wid;
  if (t >= ntok) return;
  float a0 = 0.f, a1 = 0.f, a2 = 0.f, a3 = 0.f;
  const __bf16* zr = z + (size_t)t * D_DIM;
  for (int d = lane; d < D_DIM; d += 32) {
    const float zv = (float)zr[d];
    a0 += zv * gW[d];
    a1 += zv * gW[D_DIM + d];
    a2 += zv * gW[2 * D_DIM + d];
    a3 += zv * gW[3 * D_DIM + d];
  }
#pragma unroll
  for (int o = 16; o >= 1; o >>= 1) {
    a0 += __shfl_xor(a0, o, 32);
    a1 += __shfl_xor(a1, o, 32);
    a2 += __shfl_xor(a2, o, 32);
    a3 += __shfl_xor(a3, o, 32);
  }
  if (lane == 0) {
    const float l0 = a0 + gb[0], l1 = a1 + gb[1], l2 = a2 + gb[2],
                l3 = a3 + gb[3];
    const float m = fmaxf(fmaxf(l0, l1), fmaxf(l2, l3));
    const float e0 = __expf(l0 - m), e1 = __expf(l1 - m), e2 = __expf(l2 - m),
                e3 = __expf(l3 - m);
    const float inv = 1.f / (e0 + e1 + e2 + e3);
    float4 o4 = {e0 * inv, e1 * inv, e2 * inv, e3 * inv};
    *(float4*)(gp + (size_t)t * 4) = o4;
  }
}

// ---------------------------------------------------------------------------
// f32 -> bf16 conversion (grid-stride, 4 elements/thread).
// ---------------------------------------------------------------------------
__global__ void cvt_f32_bf16(const float* __restrict__ s,
                             __bf16* __restrict__ d, long n) {
  const long i = (long)blockIdx.x * blockDim.x + threadIdx.x;
  const long stride = (long)gridDim.x * blockDim.x;
  for (long j = i * 4; j < n; j += stride * 4) {
    const float4 v = *(const float4*)(s + j);
    v4bf o = {(__bf16)v.x, (__bf16)v.y, (__bf16)v.z, (__bf16)v.w};
    *(v4bf*)(d + j) = o;
  }
}

// ---------------------------------------------------------------------------
extern "C" void kernel_launch(void* const* d_in, const int* in_sizes, int n_in,
                              void* d_out, int out_size, void* d_ws,
                              size_t ws_size, hipStream_t stream) {
  (void)n_in;
  (void)out_size;
  (void)ws_size;
  const float* x_context = (const float*)d_in[0];
  const float* x_target = (const float*)d_in[1];
  const float* enc_Win = (const float*)d_in[2];
  const float* enc_bin = (const float*)d_in[3];
  const float* enc_decay = (const float*)d_in[4];
  const float* enc_Wout = (const float*)d_in[5];
  const float* enc_bout = (const float*)d_in[6];
  const float* enc_gamma = (const float*)d_in[7];
  const float* enc_beta = (const float*)d_in[8];
  const float* tgt_Win = (const float*)d_in[9];
  const float* tgt_bin = (const float*)d_in[10];
  const float* tgt_decay = (const float*)d_in[11];
  const float* tgt_Wout = (const float*)d_in[12];
  const float* tgt_bout = (const float*)d_in[13];
  const float* tgt_gamma = (const float*)d_in[14];
  const float* tgt_beta = (const float*)d_in[15];
  const float* gate_W = (const float*)d_in[16];
  const float* gate_b = (const float*)d_in[17];
  const float* exp_W = (const float*)d_in[18];
  const float* exp_b = (const float*)d_in[19];

  const int ntok = in_sizes[0] / D_DIM;  // 8*2048 = 16384
  const long ND = (long)ntok * D_DIM;
  const long WW = (long)D_DIM * D_DIM;  // 1M elems per weight matrix

  // --- workspace carve (all 256B aligned) ---
  size_t off = 0;
  char* wsb = (char*)d_ws;
  auto take = [&](size_t bytes) -> void* {
    void* p = wsb + off;
    off = (off + bytes + 255) & ~(size_t)255;
    return p;
  };
  __bf16* wEncWin = (__bf16*)take((size_t)3 * WW * 2);
  __bf16* wEncWout = (__bf16*)take((size_t)3 * WW * 2);
  __bf16* wTgtWin = (__bf16*)take((size_t)3 * WW * 2);
  __bf16* wTgtWout = (__bf16*)take((size_t)3 * WW * 2);
  __bf16* wExp = (__bf16*)take((size_t)4 * WW * 2);
  __bf16* actA = (__bf16*)take((size_t)ND * 2);
  __bf16* actB = (__bf16*)take((size_t)ND * 2);
  __bf16* actC = (__bf16*)take((size_t)ND * 2);
  __bf16* actD = (__bf16*)take((size_t)ND * 2);
  __bf16* nsBuf = (__bf16*)take((size_t)ND * 2);
  float* tmpF = (float*)take((size_t)ND * 4);

  // --- output layout: (pred_z, gate_probs, z_target) flat ---
  float* pred = (float*)d_out;
  float* gate_out = pred + ND;
  float* ztgt_out = gate_out + (size_t)ntok * 4;

  // --- bf16 staging ---
  cvt_f32_bf16<<<1024, 256, 0, stream>>>(enc_Win, wEncWin, 3 * WW);
  cvt_f32_bf16<<<1024, 256, 0, stream>>>(enc_Wout, wEncWout, 3 * WW);
  cvt_f32_bf16<<<1024, 256, 0, stream>>>(tgt_Win, wTgtWin, 3 * WW);
  cvt_f32_bf16<<<1024, 256, 0, stream>>>(tgt_Wout, wTgtWout, 3 * WW);
  cvt_f32_bf16<<<1024, 256, 0, stream>>>(exp_W, wExp, 4 * WW);
  cvt_f32_bf16<<<2048, 256, 0, stream>>>(x_context, actA, ND);
  cvt_f32_bf16<<<2048, 256, 0, stream>>>(x_target, actC, ND);

  const dim3 ggrid(D_DIM / BN, ntok / BM);  // (8, 128)

  // --- context encoder stack ---
  __bf16* cur = actA;
  __bf16* nxt = actB;
  for (int i = 0; i < 3; ++i) {
    liquid_gemm<0><<<ggrid, 256, 0, stream>>>(
        cur, wEncWin + (size_t)i * WW, enc_bin + i * D_DIM,
        enc_decay + i * D_DIM, cur, nullptr, 0, 0, nsBuf, nullptr);
    liquid_gemm<1><<<ggrid, 256, 0, stream>>>(
        nsBuf, wEncWout + (size_t)i * WW, enc_bout + i * D_DIM, nullptr, cur,
        nullptr, 0, 0, nullptr, tmpF);
    ln_kernel<<<ntok, 256, 0, stream>>>(tmpF, enc_gamma + i * D_DIM,
                                        enc_beta + i * D_DIM, nxt, nullptr);
    __bf16* t = cur;
    cur = nxt;
    nxt = t;
  }
  __bf16* zctx = cur;  // bf16 z_context

  // --- target encoder stack (final LN also writes f32 z_target) ---
  cur = actC;
  nxt = actD;
  for (int i = 0; i < 3; ++i) {
    liquid_gemm<0><<<ggrid, 256, 0, stream>>>(
        cur, wTgtWin + (size_t)i * WW, tgt_bin + i * D_DIM,
        tgt_decay + i * D_DIM, cur, nullptr, 0, 0, nsBuf, nullptr);
    liquid_gemm<1><<<ggrid, 256, 0, stream>>>(
        nsBuf, wTgtWout + (size_t)i * WW, tgt_bout + i * D_DIM, nullptr, cur,
        nullptr, 0, 0, nullptr, tmpF);
    ln_kernel<<<ntok, 256, 0, stream>>>(tmpF, tgt_gamma + i * D_DIM,
                                        tgt_beta + i * D_DIM, nxt,
                                        (i == 2) ? ztgt_out : nullptr);
    __bf16* t = cur;
    cur = nxt;
    nxt = t;
  }

  // --- MoE predictor ---
  gate_kernel<<<(ntok + 7) / 8, 256, 0, stream>>>(zctx, gate_W, gate_b,
                                                  gate_out, ntok);
  for (int e = 0; e < 4; ++e) {
    liquid_gemm<2><<<ggrid, 256, 0, stream>>>(
        zctx, wExp + (size_t)e * WW, exp_b + e * D_DIM, nullptr, nullptr,
        gate_out, e, (e > 0) ? 1 : 0, nullptr, pred);
  }
}
